// CSBrainEncoderLayer_67319317398061
// MI455X (gfx1250) — matrix operand order repaired and checked by reading, not compiled, not run on hardware
//
#include <hip/hip_runtime.h>
#include <math.h>

// ---- problem constants ----
#define Bb   32
#define Cc   62
#define Tt   60
#define Ff   200
#define Hh   8
#define HDm  25
#define DFFm 800
#define Rr   8
#define WSm  5
#define NWm  12
#define Mtok (Bb*Cc*Tt)        // 119040, divisible by 64
#define SCL  0.2f              // 1/sqrt(25)
#define EPSm 1e-5f

typedef __bf16 v16bf __attribute__((ext_vector_type(16)));
typedef __bf16 v8bf  __attribute__((ext_vector_type(8)));
typedef float  v8f   __attribute__((ext_vector_type(8)));

#define LDSK 48   // LDS row stride (elements) = 96B -> 16B/32B aligned frag reads

// ---------------- f32 -> bf16 convert (weights, once per call) ----------------
__global__ void cvt_bf16_kernel(const float* __restrict__ in, __bf16* __restrict__ out, int n) {
  int i = blockIdx.x * blockDim.x + threadIdx.x;
  if (i < n) out[i] = (__bf16)in[i];
}

// ---------------- LayerNorm: one wave per token (F=200), f32 in -> bf16 out ----------------
__global__ __launch_bounds__(256) void ln_kernel(const float* __restrict__ x,
    const float* __restrict__ w, const float* __restrict__ bsc,
    __bf16* __restrict__ y, int ntok) {
  int lane = threadIdx.x & 31;
  int tok  = blockIdx.x * 8 + (threadIdx.x >> 5);
  if (tok >= ntok) return;
  const float* xr = x + (size_t)tok * Ff;
  float s = 0.f, ss = 0.f;
  #pragma unroll
  for (int i = 0; i < 7; ++i) {
    int d = lane + i * 32;
    if (d < Ff) { float v = xr[d]; s += v; ss += v * v; }
  }
  #pragma unroll
  for (int off = 16; off; off >>= 1) {
    s  += __shfl_xor(s,  off, 32);
    ss += __shfl_xor(ss, off, 32);
  }
  float mean = s * (1.f / Ff);
  float var  = ss * (1.f / Ff) - mean * mean;
  float rs   = rsqrtf(var + EPSm);
  __bf16* yr = y + (size_t)tok * Ff;
  #pragma unroll
  for (int i = 0; i < 7; ++i) {
    int d = lane + i * 32;
    if (d < Ff) yr[d] = (__bf16)((xr[d] - mean) * rs * w[d] + bsc[d]);
  }
}

// ---------------- WMMA GEMM, bf16 operands, async-to-LDS double buffered ----------------
// Out = A @ W^T + bias (+gelu) (+Res);  A: MxK bf16 (RMAP: row remap), W: NxK bf16.
// 64x64 tile, 128 threads = 4 waves (each a 32x32 sub-tile).
// Tiles are moved global->LDS with GLOBAL_LOAD_ASYNC_TO_LDS_B128 (ASYNCcnt),
// double-buffered so tile kt+1 is in flight while WMMA consumes tile kt.
template<int RMAP>
__global__ __launch_bounds__(128) void gemm_wmma(
    const __bf16* __restrict__ A, const int* __restrict__ rmap,
    const __bf16* __restrict__ W, const float* __restrict__ bias,
    const void* __restrict__ Res, int resBf16,
    void* __restrict__ Out, int outBf16,
    int M, int N, int K, int gelu) {
  __shared__ __align__(128) __bf16 As[2][64 * LDSK];
  __shared__ __align__(128) __bf16 Bs[2][64 * LDSK];
  int tid  = threadIdx.x;
  int lane = tid & 31;
  int wave = tid >> 5;
  int wm = wave >> 1, wn = wave & 1;
  int mBase = blockIdx.x * 64;
  int nBase = blockIdx.y * 64;

  v8f acc00 = {}, acc01 = {}, acc10 = {}, acc11 = {};
  int kTiles = (K + 31) >> 5;

  // issue async copies for K-tile kt into LDS buffer bufIdx.
  // Each thread moves 2x 16B groups per matrix (64 rows x 32 bf16 cols).
  auto issue = [&](int kt, int bufIdx) {
    int k0 = kt << 5;
    v8bf zero = {};
    #pragma unroll
    for (int p = 0; p < 2; ++p) {
      int gg   = tid + p * 128;
      int row  = gg >> 2;          // 0..63
      int col8 = (gg & 3) * 8;     // 0,8,16,24
      bool kOK = (k0 + col8 + 8) <= K;   // K % 8 == 0 here
      __bf16* lpA = &As[bufIdx][row * LDSK + col8];
      __bf16* lpB = &Bs[bufIdx][row * LDSK + col8];
      int ar = mBase + row;
      if (kOK) {
        int arow = RMAP ? rmap[ar] : ar;
        unsigned ldsA = (unsigned)(size_t)lpA;
        unsigned gofsA = (unsigned)(((size_t)arow * K + k0 + col8) * 2);
        asm volatile("global_load_async_to_lds_b128 %0, %1, %2"
                     :: "v"(ldsA), "v"(gofsA), "s"(A) : "memory");
        int nr = nBase + row;
        if (nr < N) {
          unsigned ldsB = (unsigned)(size_t)lpB;
          unsigned gofsB = (unsigned)(((size_t)nr * K + k0 + col8) * 2);
          asm volatile("global_load_async_to_lds_b128 %0, %1, %2"
                       :: "v"(ldsB), "v"(gofsB), "s"(W) : "memory");
        }
        // rows >= N are never stored (only affect discarded D columns)
      } else {
        // K padding must be zero (it feeds every output column)
        *(v8bf*)lpA = zero;
        *(v8bf*)lpB = zero;
      }
    }
  };

  issue(0, 0);
  asm volatile("s_wait_asynccnt 0x0" ::: "memory");
  __syncthreads();

  for (int kt = 0; kt < kTiles; ++kt) {
    int buf = kt & 1;
    if (kt + 1 < kTiles) issue(kt + 1, (kt + 1) & 1);   // overlap with WMMA below
    // ---- fragment loads (contiguous-K order, same for A and B) + 4x WMMA ----
    // A 16x32: lanes 0-15 row M=l, K 0-15; lanes 16-31 row M=l-16, K 16-31.
    int arow0 = wm * 32 + (lane & 15);
    int kbA   = (lane < 16) ? 0 : 16;
    union { v16bf v; v8bf h[2]; } a0u, a1u;
    a0u.h[0] = *(const v8bf*)(&As[buf][arow0 * LDSK + kbA]);
    a0u.h[1] = *(const v8bf*)(&As[buf][arow0 * LDSK + kbA + 8]);
    a1u.h[0] = *(const v8bf*)(&As[buf][(arow0 + 16) * LDSK + kbA]);
    a1u.h[1] = *(const v8bf*)(&As[buf][(arow0 + 16) * LDSK + kbA + 8]);
    // B 32x16: lanes 0-15 col N=l, K 0-15; lanes 16-31 col N=l-16, K 16-31.
    int kbB   = (lane < 16) ? 0 : 16;
    int bcol0 = wn * 32 + (lane & 15);
    v16bf b0 = *(const v16bf*)(&Bs[buf][bcol0 * LDSK + kbB]);
    v16bf b1 = *(const v16bf*)(&Bs[buf][(bcol0 + 16) * LDSK + kbB]);
    acc00 = __builtin_amdgcn_wmma_f32_16x16x32_bf16(false, a0u.v, false, b0, (short)0, acc00, false, false);
    acc01 = __builtin_amdgcn_wmma_f32_16x16x32_bf16(false, a0u.v, false, b1, (short)0, acc01, false, false);
    acc10 = __builtin_amdgcn_wmma_f32_16x16x32_bf16(false, a1u.v, false, b0, (short)0, acc10, false, false);
    acc11 = __builtin_amdgcn_wmma_f32_16x16x32_bf16(false, a1u.v, false, b1, (short)0, acc11, false, false);
    if (kt + 1 < kTiles)
      asm volatile("s_wait_asynccnt 0x0" ::: "memory");
    __syncthreads();
  }

  // D layout: VGPR p holds (M=p, N=lane) lanes 0-15; (M=8+p, N=lane-16) lanes 16-31.
  int rowSel = (lane < 16) ? 0 : 8;
  int colOff = lane & 15;
  v8f accs[2][2] = {{acc00, acc01}, {acc10, acc11}};
  #pragma unroll
  for (int i = 0; i < 2; ++i) {
    #pragma unroll
    for (int j = 0; j < 2; ++j) {
      v8f a = accs[i][j];
      int gcol = nBase + wn * 32 + j * 16 + colOff;
      if (gcol >= N) continue;
      float bv = bias ? bias[gcol] : 0.f;
      #pragma unroll
      for (int p = 0; p < 8; ++p) {
        int grow = mBase + wm * 32 + i * 16 + rowSel + p;
        if (grow >= M) continue;
        size_t oidx = (size_t)grow * N + gcol;
        float v = a[p] + bv;
        if (gelu) v = 0.5f * v * (1.f + erff(v * 0.70710678118f));
        if (Res) v += resBf16 ? (float)((const __bf16*)Res)[oidx]
                              : ((const float*)Res)[oidx];
        if (outBf16) ((__bf16*)Out)[oidx] = (__bf16)v;
        else         ((float*)Out)[oidx]  = v;
      }
    }
  }
}

// ---------------- window attention: block per (b,c,s); thread per (head, query) ----------------
__global__ __launch_bounds__(96) void win_attn_kernel(const __bf16* __restrict__ qkv,
                                                      __bf16* __restrict__ out) {
  __shared__ float lds[NWm][3 * Ff];   // 12 x 600 f32 = 28.8 KB
  int blk = blockIdx.x;
  int s = blk % WSm;
  int c = (blk / WSm) % Cc;
  int b = blk / (WSm * Cc);
  int tid = threadIdx.x;
  size_t tokBase = ((size_t)(b * Cc + c)) * Tt + s;
  for (int idx = tid; idx < NWm * 3 * Ff; idx += 96) {
    int w = idx / (3 * Ff);
    int col = idx % (3 * Ff);
    lds[w][col] = (float)qkv[(tokBase + (size_t)w * WSm) * (3 * Ff) + col];
  }
  __syncthreads();
  int h  = tid / NWm;   // 0..7
  int wq = tid % NWm;   // 0..11
  float q[HDm], o[HDm];
  #pragma unroll
  for (int d = 0; d < HDm; ++d) { q[d] = lds[wq][h * HDm + d]; o[d] = 0.f; }
  float m = -3.0e38f, l = 0.f;
  for (int wk = 0; wk < NWm; ++wk) {
    float sc = 0.f;
    #pragma unroll
    for (int d = 0; d < HDm; ++d) sc += q[d] * lds[wk][Ff + h * HDm + d];
    sc *= SCL;
    float mN = fmaxf(m, sc);
    float corr = __expf(m - mN);
    float p = __expf(sc - mN);
    l = l * corr + p;
    #pragma unroll
    for (int d = 0; d < HDm; ++d) o[d] = o[d] * corr + p * lds[wk][2 * Ff + h * HDm + d];
    m = mN;
  }
  float inv = 1.f / l;
  size_t tok = tokBase + (size_t)wq * WSm;
  #pragma unroll
  for (int d = 0; d < HDm; ++d) out[tok * Ff + h * HDm + d] = (__bf16)(o[d] * inv);
}

// ---------------- region attention: block per (b,t,h); thread per query channel ----------------
__global__ __launch_bounds__(64) void reg_attn_kernel(const __bf16* __restrict__ qkv,
                                                      __bf16* __restrict__ out) {
  __shared__ float Ks[Cc][HDm];
  __shared__ float Vs[Cc][HDm];
  int blk = blockIdx.x;
  int h = blk % Hh;
  int t = (blk / Hh) % Tt;
  int b = blk / (Hh * Tt);
  int tid = threadIdx.x;
  for (int idx = tid; idx < Cc * HDm; idx += 64) {
    int ck = idx / HDm, d = idx % HDm;
    size_t tok = ((size_t)(b * Cc + ck)) * Tt + t;
    Ks[ck][d] = (float)qkv[tok * (3 * Ff) + Ff + h * HDm + d];
    Vs[ck][d] = (float)qkv[tok * (3 * Ff) + 2 * Ff + h * HDm + d];
  }
  __syncthreads();
  int qi = tid;
  if (qi >= Cc) return;
  size_t qtok = ((size_t)(b * Cc + qi)) * Tt + t;
  float q[HDm], o[HDm];
  #pragma unroll
  for (int d = 0; d < HDm; ++d) { q[d] = (float)qkv[qtok * (3 * Ff) + h * HDm + d]; o[d] = 0.f; }
  float m = -3.0e38f, l = 0.f;
  for (int ck = 0; ck < Cc; ++ck) {
    float sc = 0.f;
    #pragma unroll
    for (int d = 0; d < HDm; ++d) sc += q[d] * Ks[ck][d];
    sc *= SCL;
    float mN = fmaxf(m, sc);
    float corr = __expf(m - mN);
    float p = __expf(sc - mN);
    l = l * corr + p;
    #pragma unroll
    for (int d = 0; d < HDm; ++d) o[d] = o[d] * corr + p * Vs[ck][d];
    m = mN;
  }
  float inv = 1.f / l;
  #pragma unroll
  for (int d = 0; d < HDm; ++d) out[qtok * Ff + h * HDm + d] = (__bf16)(o[d] * inv);
}

// ---------------- region means over channel groups (bf16 in/out) ----------------
__global__ void reg_means_kernel(const __bf16* __restrict__ xln, __bf16* __restrict__ means) {
  int idx = blockIdx.x * blockDim.x + threadIdx.x;
  int total = Bb * Rr * Tt * Ff;
  if (idx >= total) return;
  int f = idx % Ff;
  int t = (idx / Ff) % Tt;
  int r = (idx / (Ff * Tt)) % Rr;
  int b = idx / (Ff * Tt * Rr);
  int c0 = r * 8;
  int cnt = (c0 + 8 <= Cc) ? 8 : (Cc - c0);
  float sum = 0.f;
  for (int c = 0; c < cnt; ++c)
    sum += (float)xln[(((size_t)(b * Cc + c0 + c)) * Tt + t) * Ff + f];
  means[idx] = (__bf16)(sum / (float)cnt);
}

// ---------------- token -> region-mean-row map ----------------
__global__ void rowmap_kernel(const int* __restrict__ rid, int* __restrict__ map) {
  int tok = blockIdx.x * blockDim.x + threadIdx.x;
  if (tok >= Mtok) return;
  int t = tok % Tt;
  int c = (tok / Tt) % Cc;
  int b = tok / (Tt * Cc);
  map[tok] = (b * Rr + rid[c]) * Tt + t;
}

extern "C" void kernel_launch(void* const* d_in, const int* in_sizes, int n_in,
                              void* d_out, int out_size, void* d_ws, size_t ws_size,
                              hipStream_t stream) {
  const float* src       = (const float*)d_in[0];
  const int*   rid       = (const int*)  d_in[1];
  const float* win_in_w  = (const float*)d_in[2];
  const float* win_in_b  = (const float*)d_in[3];
  const float* win_out_w = (const float*)d_in[4];
  const float* win_out_b = (const float*)d_in[5];
  const float* reg_in_w  = (const float*)d_in[6];
  const float* reg_in_b  = (const float*)d_in[7];
  const float* reg_out_w = (const float*)d_in[8];
  const float* reg_out_b = (const float*)d_in[9];
  const float* gfc_w     = (const float*)d_in[10];
  const float* gfc_b     = (const float*)d_in[11];
  const float* lin1_w    = (const float*)d_in[12];
  const float* lin1_b    = (const float*)d_in[13];
  const float* lin2_w    = (const float*)d_in[14];
  const float* lin2_b    = (const float*)d_in[15];
  const float* ln1_w     = (const float*)d_in[16];
  const float* ln1_b     = (const float*)d_in[17];
  const float* ln2_w     = (const float*)d_in[18];
  const float* ln2_b     = (const float*)d_in[19];
  const float* ln3_w     = (const float*)d_in[20];
  const float* ln3_b     = (const float*)d_in[21];
  float* x = (float*)d_out;

  // ---- workspace layout (bf16 activations) ----
  __bf16* bufLN    = (__bf16*)d_ws;                          // M x 200
  __bf16* bufO     = bufLN    + (size_t)Mtok * Ff;           // M x 200
  __bf16* bufQKV   = bufO     + (size_t)Mtok * Ff;           // M x 600
  __bf16* bufH     = bufQKV   + (size_t)Mtok * 3 * Ff;       // M x 800
  __bf16* bufMeans = bufH     + (size_t)Mtok * DFFm;         // (B*R*T) x 200
  __bf16* wWinIn   = bufMeans + (size_t)Bb * Rr * Tt * Ff;   // weights (bf16)
  __bf16* wWinOut  = wWinIn   + 3 * Ff * Ff;
  __bf16* wRegIn   = wWinOut  + Ff * Ff;
  __bf16* wRegOut  = wRegIn   + 3 * Ff * Ff;
  __bf16* wGfc     = wRegOut  + Ff * Ff;
  __bf16* wLin1    = wGfc     + Ff * Ff;
  __bf16* wLin2    = wLin1    + DFFm * Ff;
  int*    rowmap   = (int*)(wLin2 + Ff * DFFm);

  dim3 blk128(128);
  int mT = Mtok / 64;   // 1860 tiles

  // ---- weight conversions (tiny, once per call) ----
  cvt_bf16_kernel<<<(3*Ff*Ff + 255)/256, 256, 0, stream>>>(win_in_w, wWinIn, 3*Ff*Ff);
  cvt_bf16_kernel<<<(Ff*Ff   + 255)/256, 256, 0, stream>>>(win_out_w, wWinOut, Ff*Ff);
  cvt_bf16_kernel<<<(3*Ff*Ff + 255)/256, 256, 0, stream>>>(reg_in_w, wRegIn, 3*Ff*Ff);
  cvt_bf16_kernel<<<(Ff*Ff   + 255)/256, 256, 0, stream>>>(reg_out_w, wRegOut, Ff*Ff);
  cvt_bf16_kernel<<<(Ff*Ff   + 255)/256, 256, 0, stream>>>(gfc_w, wGfc, Ff*Ff);
  cvt_bf16_kernel<<<(DFFm*Ff + 255)/256, 256, 0, stream>>>(lin1_w, wLin1, DFFm*Ff);
  cvt_bf16_kernel<<<(Ff*DFFm + 255)/256, 256, 0, stream>>>(lin2_w, wLin2, Ff*DFFm);
  rowmap_kernel<<<(Mtok + 255)/256, 256, 0, stream>>>(rid, rowmap);

  // --- sublayer 1: window attention ---
  ln_kernel<<<(Mtok + 7) / 8, 256, 0, stream>>>(src, ln1_w, ln1_b, bufLN, Mtok);
  gemm_wmma<0><<<dim3(mT, (3 * Ff + 63) / 64), blk128, 0, stream>>>(
      bufLN, nullptr, wWinIn, win_in_b, nullptr, 0, bufQKV, 1, Mtok, 3 * Ff, Ff, 0);
  win_attn_kernel<<<Bb * Cc * WSm, 96, 0, stream>>>(bufQKV, bufO);
  gemm_wmma<0><<<dim3(mT, (Ff + 63) / 64), blk128, 0, stream>>>(
      bufO, nullptr, wWinOut, win_out_b, src, 0, x, 0, Mtok, Ff, Ff, 0);

  // --- sublayer 2: region attention ---
  ln_kernel<<<(Mtok + 7) / 8, 256, 0, stream>>>(x, ln2_w, ln2_b, bufLN, Mtok);
  {
    int total = Bb * Rr * Tt * Ff;
    reg_means_kernel<<<(total + 255) / 256, 256, 0, stream>>>(bufLN, bufMeans);
  }
  // xe = xln2 + gf @ gfc_w^T + gfc_b   (A = region means, row-remapped per token)
  gemm_wmma<1><<<dim3(mT, (Ff + 63) / 64), blk128, 0, stream>>>(
      bufMeans, rowmap, wGfc, gfc_b, bufLN, 1, bufO, 1, Mtok, Ff, Ff, 0);
  gemm_wmma<0><<<dim3(mT, (3 * Ff + 63) / 64), blk128, 0, stream>>>(
      bufO, nullptr, wRegIn, reg_in_b, nullptr, 0, bufQKV, 1, Mtok, 3 * Ff, Ff, 0);
  reg_attn_kernel<<<Bb * Tt * Hh, 64, 0, stream>>>(bufQKV, bufLN);
  gemm_wmma<0><<<dim3(mT, (Ff + 63) / 64), blk128, 0, stream>>>(
      bufLN, nullptr, wRegOut, reg_out_b, x, 0, x, 0, Mtok, Ff, Ff, 0);

  // --- sublayer 3: FFN ---
  ln_kernel<<<(Mtok + 7) / 8, 256, 0, stream>>>(x, ln3_w, ln3_b, bufLN, Mtok);
  gemm_wmma<0><<<dim3(mT, (DFFm + 63) / 64), blk128, 0, stream>>>(
      bufLN, nullptr, wLin1, lin1_b, nullptr, 0, bufH, 1, Mtok, DFFm, Ff, 1);
  gemm_wmma<0><<<dim3(mT, (Ff + 63) / 64), blk128, 0, stream>>>(
      bufH, nullptr, wLin2, lin2_b, x, 0, x, 0, Mtok, Ff, DFFm, 0);
}
